// PatchMerging3D_52321291600483
// MI455X (gfx1250) — matrix operand order repaired and checked by reading, not compiled, not run on hardware
//
#include <hip/hip_runtime.h>
#include <hip/hip_bf16.h>

typedef __attribute__((ext_vector_type(16))) __bf16 v16bf;
typedef __attribute__((ext_vector_type(8)))  __bf16 v8bf;
typedef __attribute__((ext_vector_type(2)))  __bf16 v2bf;
typedef __attribute__((ext_vector_type(8)))  float  v8f;
typedef __attribute__((ext_vector_type(4)))  float  v4f;

#define CC   96      // C
#define K8   768     // 8*C (LN width / GEMM K)
#define NO   192     // 2*C (GEMM N)
#define LB   6272    // merged rows per batch (8*28*28)
#define MT   64      // merged rows per block
#define LDA  776     // padded LDS row stride in bf16 elems (+4 banks/row -> conflict-free)

// ---- prep: f32 weights -> bf16 in workspace (same [192][768] row-major layout) ----
__global__ void pm3d_wcvt(const float* __restrict__ w, __bf16* __restrict__ wbf, int n) {
    int i = blockIdx.x * blockDim.x + threadIdx.x;
    if (i < n) wbf[i] = (__bf16)w[i];
}

// ---- fused merge + layernorm + GEMM ----
__global__ __launch_bounds__(256)
void pm3d_kernel(const float* __restrict__ x,
                 const float* __restrict__ gamma,
                 const float* __restrict__ beta,
                 const __bf16* __restrict__ wbf,
                 float* __restrict__ out) {
    __shared__ __bf16 As[MT * LDA];   // 64 x 768 bf16 (padded) ~ 97 KB

    const int tid   = threadIdx.x;
    const int grow0 = blockIdx.x * MT;

    // ================= Phase 1: merge + layernorm -> bf16 LDS =================
    {
        const int r    = tid >> 2;      // row within block: 0..63
        const int sub  = tid & 3;       // 4 threads per row
        const int grow = grow0 + r;
        const int b    = grow / LB;
        const int l    = grow - b * LB;
        const int d2   = l / 784;
        const int t0   = l - d2 * 784;
        const int h2   = t0 / 28;
        const int w2   = t0 - h2 * 28;

        const float* xb = x + (size_t)b * (16 * 56 * 56) * CC;

        float sum = 0.f, ssq = 0.f;
        #pragma unroll
        for (int cj = 0; cj < 2; ++cj) {
            const int j  = sub + cj * 4;               // chunk 0..7 = (db,hb,wb)
            const int db = j >> 2, hb = (j >> 1) & 1, wb = j & 1;
            const size_t sl = (size_t)((2 * d2 + db) * 3136 + (2 * h2 + hb) * 56 + (2 * w2 + wb));
            const float* src = xb + sl * CC;
            for (int c = 0; c < CC; c += 4) {
                v4f v = *(const v4f*)(src + c);
                sum += v.x + v.y + v.z + v.w;
                ssq += v.x * v.x + v.y * v.y + v.z * v.z + v.w * v.w;
            }
        }
        // reduce across the 4 lanes owning this row (consecutive lanes in a wave32)
        sum += __shfl_xor(sum, 1, 32);  ssq += __shfl_xor(ssq, 1, 32);
        sum += __shfl_xor(sum, 2, 32);  ssq += __shfl_xor(ssq, 2, 32);

        const float mean = sum * (1.f / 768.f);
        const float var  = ssq * (1.f / 768.f) - mean * mean;
        const float rstd = rsqrtf(var + 1e-5f);

        #pragma unroll
        for (int cj = 0; cj < 2; ++cj) {
            const int j  = sub + cj * 4;
            const int db = j >> 2, hb = (j >> 1) & 1, wb = j & 1;
            const size_t sl = (size_t)((2 * d2 + db) * 3136 + (2 * h2 + hb) * 56 + (2 * w2 + wb));
            const float* src = xb + sl * CC;            // L2 hit (just streamed)
            const float* g   = gamma + j * CC;
            const float* bt  = beta  + j * CC;
            __bf16* dst = &As[r * LDA + j * CC];
            for (int c = 0; c < CC; c += 2) {
                float v0 = (src[c]     - mean) * rstd * g[c]     + bt[c];
                float v1 = (src[c + 1] - mean) * rstd * g[c + 1] + bt[c + 1];
                v2bf p; p.x = (__bf16)v0; p.y = (__bf16)v1;
                *(v2bf*)(dst + c) = p;                  // 4B-aligned ds_store
            }
        }
    }
    __syncthreads();

    // ================= Phase 2: GEMM 64x192 += A(64x768 bf16) * W^T =================
    const int wave  = tid >> 5;        // 0..7
    const int lane  = tid & 31;
    const int strip = wave & 3;        // 16-row strip
    const int nhalf = wave >> 2;       // 0/1 -> 96-col half
    const int m     = lane & 15;       // M (A) / N (B,C) index for this lane
    const int half  = lane >> 4;       // lane half selects K/M split per ISA layout

    v8f acc[6] = {};

    // Invariant base pointers: all K-step addressing becomes immediate offsets.
    // A 16x32 bf16 layout: lane(m,half): V0-7 <- K=k0+half*16..+15 (contiguous)
    const __bf16* abase = &As[(strip * 16 + m) * LDA] + half * 16;
    // B 32x16 bf16 layout: lane(n,half): V0-7 <- column n, K=k0+half*16..+15 (contiguous)
    const __bf16* bbase[6];
    #pragma unroll
    for (int t = 0; t < 6; ++t)
        bbase[t] = wbf + (size_t)(nhalf * 96 + t * 16 + m) * K8 + half * 16;

    #pragma unroll
    for (int kk = 0; kk < 24; ++kk) {
        const int k0 = kk * 32;                       // compile-time in unrolled body
        v8bf alo = *(const v8bf*)(abase + k0);        // ds_load_b128, imm offset
        v8bf ahi = *(const v8bf*)(abase + k0 + 8);
        v16bf a  = __builtin_shufflevector(alo, ahi,
                    0, 1, 2, 3, 4, 5, 6, 7, 8, 9, 10, 11, 12, 13, 14, 15);

        // Stage all 6 B fragments first so the 12 global_load_b128 issue as a
        // clause and drain with descending loadcnt while WMMAs overlap.
        v16bf bf[6];
        #pragma unroll
        for (int t = 0; t < 6; ++t)
            bf[t] = *(const v16bf*)(bbase[t] + k0);   // imm offset kk*64 bytes

        #pragma unroll
        for (int t = 0; t < 6; ++t)
            acc[t] = __builtin_amdgcn_wmma_f32_16x16x32_bf16(
                         false, a, false, bf[t], (short)0, acc[t], false, false);
    }

    // C/D layout: VGPR i, lanes 0-15 -> M=i, lanes 16-31 -> M=i+8; N = lane&15
    #pragma unroll
    for (int t = 0; t < 6; ++t) {
        const int n = nhalf * 96 + t * 16 + m;
        #pragma unroll
        for (int i = 0; i < 8; ++i) {
            const int row = strip * 16 + (half ? i + 8 : i);
            out[(size_t)(grow0 + row) * NO + n] = acc[t][i];
        }
    }
}

extern "C" void kernel_launch(void* const* d_in, const int* in_sizes, int n_in,
                              void* d_out, int out_size, void* d_ws, size_t ws_size,
                              hipStream_t stream) {
    const float* x     = (const float*)d_in[0];
    const float* gamma = (const float*)d_in[1];
    const float* beta  = (const float*)d_in[2];
    const float* w     = (const float*)d_in[3];
    float*       out   = (float*)d_out;
    __bf16*      wbf   = (__bf16*)d_ws;   // 192*768*2 = 294,912 B of scratch

    const int nw = NO * K8;
    pm3d_wcvt<<<(nw + 255) / 256, 256, 0, stream>>>(w, wbf, nw);

    const int nrows = 8 * LB;             // 50176 merged rows
    pm3d_kernel<<<nrows / MT, 256, 0, stream>>>(x, gamma, beta, wbf, out);
}
